// NestedModuleTokenizer_74972949119347
// MI455X (gfx1250) — matrix operand order repaired and checked by reading, not compiled, not run on hardware
//
#include <hip/hip_runtime.h>
#include <hip/hip_bf16.h>

typedef float v8f __attribute__((ext_vector_type(8)));
typedef __bf16 v16bf __attribute__((ext_vector_type(16)));
typedef unsigned int v4u __attribute__((ext_vector_type(4)));

#define MTOK 4096   // B*N tokens
#define EMBD 1024
#define NSEQ 1024
#define NBAT 4
#define NHEAD 16
#define DHEAD 64

// ---------------- bf16 split helpers ----------------
__device__ __forceinline__ unsigned short f2bf_rn(float f) {
  unsigned int u = __float_as_uint(f);
  unsigned int lsb = (u >> 16) & 1u;
  u += 0x7fffu + lsb;                 // round-to-nearest-even
  return (unsigned short)(u >> 16);
}
__device__ __forceinline__ float bf2f(unsigned short h) {
  return __uint_as_float(((unsigned int)h) << 16);
}

// x -> hi + lo (bf16 pair), row-major, elementwise
__global__ __launch_bounds__(256) void split_bf16_kernel(const float* __restrict__ src,
                                                         unsigned short* __restrict__ hi,
                                                         unsigned short* __restrict__ lo,
                                                         int total) {
  int i = blockIdx.x * 256 + threadIdx.x;
  if (i >= total) return;
  float x = src[i];
  unsigned short h = f2bf_rn(x);
  float r = x - bf2f(h);
  hi[i] = h;
  lo[i] = f2bf_rn(r);
}

// weight W (K x N, row-major fp32) -> Wt_hi/Wt_lo (N x K, bf16) for contiguous B fragments
__global__ __launch_bounds__(256) void splitT_bf16_kernel(const float* __restrict__ src,
                                                          unsigned short* __restrict__ hi,
                                                          unsigned short* __restrict__ lo,
                                                          int K, int N) {
  int i = blockIdx.x * 256 + threadIdx.x;   // over K*N (exact multiple of 256)
  int k = i / N, n = i - k * N;
  float x = src[i];
  unsigned short h = f2bf_rn(x);
  float r = x - bf2f(h);
  size_t o = (size_t)n * K + k;
  hi[o] = h;
  lo[o] = f2bf_rn(r);
}

// ---------------- gating ----------------
__global__ __launch_bounds__(256) void gate_kernel(const int* __restrict__ idx,
                                                   const float* __restrict__ sw,
                                                   float* __restrict__ we,
                                                   float* __restrict__ sel) {
  int t = blockIdx.x * 256 + threadIdx.x;
  if (t >= MTOK) return;
  int i0 = idx[2 * t], i1 = idx[2 * t + 1];
  float w0 = sw[2 * t], w1 = sw[2 * t + 1];
  #pragma unroll
  for (int e = 0; e < 8; ++e) {
    float w = 0.f, s = 0.f;
    if (i0 == e) { w += w0; s = 1.f; }
    if (i1 == e) { w += w1; s = 1.f; }
    we[e * MTOK + t] = w;
    sel[e * MTOK + t] = s;
  }
}

// y = (we0 + we1) * x   (identity experts folded into init)
__global__ __launch_bounds__(256) void init_y_kernel(float* __restrict__ y,
                                                     const float* __restrict__ x,
                                                     const float* __restrict__ we) {
  int i = blockIdx.x * 256 + threadIdx.x;
  int t = i >> 10;
  y[i] = (we[t] + we[MTOK + t]) * x[i];
}

// y += we[t] * (base + delta)
__global__ __launch_bounds__(256) void accum_kernel(float* __restrict__ y,
                                                    const float* __restrict__ we,
                                                    const float* __restrict__ base,
                                                    const float* __restrict__ delta) {
  int i = blockIdx.x * 256 + threadIdx.x;
  int t = i >> 10;
  y[i] += we[t] * (base[i] + delta[i]);
}

// h = x + p
__global__ __launch_bounds__(256) void add_kernel(float* __restrict__ h,
                                                  const float* __restrict__ x,
                                                  const float* __restrict__ p) {
  int i = blockIdx.x * 256 + threadIdx.x;
  h[i] = x[i] + p[i];
}

// ---------------- layernorm ----------------
__global__ __launch_bounds__(256) void ln_kernel(const float* __restrict__ x,
                                                 const float* __restrict__ g,
                                                 const float* __restrict__ b,
                                                 float* __restrict__ out) {
  __shared__ float red[256];
  int row = blockIdx.x;
  int tid = threadIdx.x;
  const float* xr = x + (size_t)row * EMBD;
  float s = 0.f;
  for (int i = tid; i < EMBD; i += 256) s += xr[i];
  red[tid] = s; __syncthreads();
  for (int k = 128; k > 0; k >>= 1) { if (tid < k) red[tid] += red[tid + k]; __syncthreads(); }
  float mu = red[0] * (1.f / EMBD);
  __syncthreads();
  float v = 0.f;
  for (int i = tid; i < EMBD; i += 256) { float d = xr[i] - mu; v += d * d; }
  red[tid] = v; __syncthreads();
  for (int k = 128; k > 0; k >>= 1) { if (tid < k) red[tid] += red[tid + k]; __syncthreads(); }
  float rstd = rsqrtf(red[0] * (1.f / EMBD) + 1e-5f);
  float* orow = out + (size_t)row * EMBD;
  for (int i = tid; i < EMBD; i += 256) orow[i] = g[i] * (xr[i] - mu) * rstd + b[i];
}

// ---------------- bf16x3 WMMA GEMM with LDS-staged B ----------------
// C = A(MxK) @ W(KxN) + bias. A: row-major bf16 hi/lo. W: N-major bf16 hi/lo.
// Block 256 = 8 waves -> 256(M) x 64(N) C region; wave -> 32x64 (2 m-tiles x 4 n-tiles).
// Per K-chunk 32: block stages the 64-col x 32-K hi/lo B tile into LDS (8KB) once,
// shared by all 8 waves; each wave issues 24 v_wmma_f32_16x16x32_bf16.
// 16-bit A layout (ISA 7.12.2): lane r holds K {8hh..+7} then {16+8hh..+7}.
// 16-bit B layout: lane r = column, K {8hh..+7} then {16+8hh..+7} (matches A).
__global__ __launch_bounds__(256) void wmma_gemm_bf16x3(const unsigned short* __restrict__ Ah,
                                                        const unsigned short* __restrict__ Al,
                                                        const unsigned short* __restrict__ Bh,
                                                        const unsigned short* __restrict__ Bl,
                                                        const float* __restrict__ bias,
                                                        float* __restrict__ C,
                                                        int K, int N) {
  union Frag { v16bf v; v4u u[2]; };
  __shared__ unsigned short ldsB[2][64][32];   // [hi/lo][col][k-in-chunk]
  int tid = threadIdx.x;
  int lane = tid & 31;
  int wid  = tid >> 5;
  int r  = lane & 15;
  int hh = lane >> 4;
  int n0 = blockIdx.x * 64;
  int m0 = blockIdx.y * 256 + wid * 32;

  v8f acc[2][4] = {};
  const unsigned short* A0H = Ah + (size_t)(m0 + r) * K + 8 * hh;
  const unsigned short* A0L = Al + (size_t)(m0 + r) * K + 8 * hh;
  const unsigned short* A1H = A0H + (size_t)16 * K;
  const unsigned short* A1L = A0L + (size_t)16 * K;

  for (int k = 0; k < K; k += 32) {
    __syncthreads();   // previous chunk's readers done before overwrite
    // stage B tile: 512 x 16B chunks, 2 per thread
    #pragma unroll
    for (int rep = 0; rep < 2; ++rep) {
      int c16 = rep * 256 + tid;
      int mtx = c16 >> 8;          // 0 = hi, 1 = lo
      int rem = c16 & 255;
      int col = rem >> 2;
      int part = rem & 3;
      const unsigned short* src =
          (mtx ? Bl : Bh) + (size_t)(n0 + col) * K + k + part * 8;
      *(v4u*)&ldsB[mtx][col][part * 8] = *(const v4u*)src;
    }
    __syncthreads();

    Frag a0h, a0l, a1h, a1l;
    a0h.u[0] = *(const v4u*)(A0H + k);  a0h.u[1] = *(const v4u*)(A0H + k + 16);
    a0l.u[0] = *(const v4u*)(A0L + k);  a0l.u[1] = *(const v4u*)(A0L + k + 16);
    a1h.u[0] = *(const v4u*)(A1H + k);  a1h.u[1] = *(const v4u*)(A1H + k + 16);
    a1l.u[0] = *(const v4u*)(A1L + k);  a1l.u[1] = *(const v4u*)(A1L + k + 16);

    #pragma unroll
    for (int t = 0; t < 4; ++t) {
      Frag bhf, blf;
      const unsigned short* ph = &ldsB[0][t * 16 + r][8 * hh];
      const unsigned short* pl = &ldsB[1][t * 16 + r][8 * hh];
      bhf.u[0] = *(const v4u*)(ph);  bhf.u[1] = *(const v4u*)(ph + 16);
      blf.u[0] = *(const v4u*)(pl);  blf.u[1] = *(const v4u*)(pl + 16);
      acc[0][t] = __builtin_amdgcn_wmma_f32_16x16x32_bf16(false, a0h.v, false, bhf.v, (short)0, acc[0][t], false, false);
      acc[0][t] = __builtin_amdgcn_wmma_f32_16x16x32_bf16(false, a0h.v, false, blf.v, (short)0, acc[0][t], false, false);
      acc[0][t] = __builtin_amdgcn_wmma_f32_16x16x32_bf16(false, a0l.v, false, bhf.v, (short)0, acc[0][t], false, false);
      acc[1][t] = __builtin_amdgcn_wmma_f32_16x16x32_bf16(false, a1h.v, false, bhf.v, (short)0, acc[1][t], false, false);
      acc[1][t] = __builtin_amdgcn_wmma_f32_16x16x32_bf16(false, a1h.v, false, blf.v, (short)0, acc[1][t], false, false);
      acc[1][t] = __builtin_amdgcn_wmma_f32_16x16x32_bf16(false, a1l.v, false, bhf.v, (short)0, acc[1][t], false, false);
    }
  }

  float bi0 = bias[n0 + r];
  float bi1 = bias[n0 + r + 16];
  float bi2 = bias[n0 + r + 32];
  float bi3 = bias[n0 + r + 48];
  #pragma unroll
  for (int mt = 0; mt < 2; ++mt) {
    #pragma unroll
    for (int i = 0; i < 8; ++i) {
      float* Crow = C + (size_t)(m0 + mt * 16 + i + 8 * hh) * N + n0 + r;  // C VGPR i -> rows i / i+8
      Crow[0]  = acc[mt][0][i] + bi0;
      Crow[16] = acc[mt][1][i] + bi1;
      Crow[32] = acc[mt][2][i] + bi2;
      Crow[48] = acc[mt][3][i] + bi3;
    }
  }
}

// ---------------- gelu (tanh approximation, jax.nn.gelu default) ----------------
__global__ __launch_bounds__(256) void gelu_kernel(float* __restrict__ h, int total) {
  int i = blockIdx.x * 256 + threadIdx.x;
  if (i >= total) return;
  float x = h[i];
  float c = 0.7978845608028654f * (x + 0.044715f * x * x * x);
  h[i] = 0.5f * x * (1.f + tanhf(c));
}

// ---------------- attention: one block per (b, head, query row) ----------------
__global__ __launch_bounds__(256) void attn_kernel(const float* __restrict__ qkv,
                                                   const float* __restrict__ sel,
                                                   float* __restrict__ out) {
  __shared__ float qs[DHEAD];
  __shared__ float sc[NSEQ];
  __shared__ float red[256];
  int i = blockIdx.x;
  int hd = blockIdx.y;
  int b = blockIdx.z;
  int tid = threadIdx.x;

  const float* base = qkv + (size_t)(b * NSEQ) * (3 * EMBD);
  const float* qrow = base + (size_t)i * (3 * EMBD) + hd * DHEAD;
  if (tid < DHEAD) qs[tid] = qrow[tid];
  __syncthreads();

  for (int j = tid; j < NSEQ; j += 256) {
    const float* krow = base + (size_t)j * (3 * EMBD) + EMBD + hd * DHEAD;
    float d = 0.f;
    #pragma unroll 8
    for (int t = 0; t < DHEAD; ++t) d += qs[t] * krow[t];
    bool allow = (sel[b * NSEQ + j] != 0.f) || (j == i);
    sc[j] = allow ? d * 0.125f : -1.0e9f;
  }
  __syncthreads();

  float m = -3.4e38f;
  for (int j = tid; j < NSEQ; j += 256) m = fmaxf(m, sc[j]);
  red[tid] = m; __syncthreads();
  for (int k = 128; k > 0; k >>= 1) { if (tid < k) red[tid] = fmaxf(red[tid], red[tid + k]); __syncthreads(); }
  m = red[0];
  __syncthreads();

  float s = 0.f;
  for (int j = tid; j < NSEQ; j += 256) { float e = expf(sc[j] - m); sc[j] = e; s += e; }
  red[tid] = s; __syncthreads();
  for (int k = 128; k > 0; k >>= 1) { if (tid < k) red[tid] += red[tid + k]; __syncthreads(); }
  float inv = 1.f / red[0];
  __syncthreads();

  int d = tid & 63, c = tid >> 6;
  float o = 0.f;
  for (int j = c * 256; j < (c + 1) * 256; ++j) {
    const float* vrow = base + (size_t)j * (3 * EMBD) + 2 * EMBD + hd * DHEAD;
    o += sc[j] * vrow[d];
  }
  red[tid] = o; __syncthreads();
  if (c == 0) {
    o = red[d] + red[d + 64] + red[d + 128] + red[d + 192];
    out[(size_t)(b * NSEQ + i) * EMBD + hd * DHEAD + d] = o * inv;
  }
}

// ---------------- host orchestration ----------------
extern "C" void kernel_launch(void* const* d_in, const int* in_sizes, int n_in,
                              void* d_out, int out_size, void* d_ws, size_t ws_size,
                              hipStream_t stream) {
  const float* x  = (const float*)d_in[0];
  const float* sw = (const float*)d_in[1];
  const int*  idx = (const int*)d_in[50];
  float* y = (float*)d_out;

  auto P = [&](int i) { return (const float*)d_in[i]; };

  // workspace layout
  float* ws    = (float*)d_ws;
  float* we    = ws;                                  // 8*MTOK
  float* sel   = ws + 8 * MTOK;                       // 8*MTOK
  float* bln   = ws + 16 * MTOK;                      // MTOK*EMBD
  float* bqkv  = bln  + (size_t)MTOK * EMBD;          // 3*MTOK*EMBD
  float* bctx  = bqkv + (size_t)3 * MTOK * EMBD;      // MTOK*EMBD
  float* bproj = bctx + (size_t)MTOK * EMBD;          // MTOK*EMBD
  float* bh    = bproj + (size_t)MTOK * EMBD;         // MTOK*EMBD
  float* bhid  = bh   + (size_t)MTOK * EMBD;          // 4*MTOK*EMBD
  unsigned short* a_hi  = (unsigned short*)(bhid + (size_t)4 * MTOK * EMBD); // MTOK*4E
  unsigned short* a_lo  = a_hi + (size_t)MTOK * 4 * EMBD;
  unsigned short* wt_hi = a_lo + (size_t)MTOK * 4 * EMBD;                    // 4E*E
  unsigned short* wt_lo = wt_hi + (size_t)4 * EMBD * EMBD;

  const int EW = MTOK * EMBD / 256;
  gate_kernel<<<MTOK / 256, 256, 0, stream>>>(idx, sw, we, sel);
  init_y_kernel<<<EW, 256, 0, stream>>>(y, x, we);

  // C = A @ W + bias via bf16x3 WMMA; converts A and W each call.
  auto gemm = [&](const float* A, const float* W, const float* bias, float* Cout, int K, int N) {
    split_bf16_kernel<<<(MTOK * K) / 256, 256, 0, stream>>>(A, a_hi, a_lo, MTOK * K);
    splitT_bf16_kernel<<<(K * N) / 256, 256, 0, stream>>>(W, wt_hi, wt_lo, K, N);
    dim3 g(N / 64, MTOK / 256);
    wmma_gemm_bf16x3<<<g, 256, 0, stream>>>(a_hi, a_lo, wt_hi, wt_lo, bias, Cout, K, N);
  };
  auto ln = [&](const float* in, const float* g_, const float* b_, float* o) {
    ln_kernel<<<MTOK, 256, 0, stream>>>(in, g_, b_, o);
  };

  auto attn_block = [&](const float* src, int e,
                        const float* wqkv, const float* bqkvp,
                        const float* wo, const float* bo,
                        const float* lng, const float* lnb) {
    ln(src, lng, lnb, bln);
    gemm(bln, wqkv, bqkvp, bqkv, EMBD, 3 * EMBD);
    attn_kernel<<<dim3(NSEQ, NHEAD, NBAT), 256, 0, stream>>>(bqkv, sel + e * MTOK, bctx);
    gemm(bctx, wo, bo, bproj, EMBD, EMBD);
  };
  auto mlp_block = [&](const float* src,
                       const float* w1, const float* b1,
                       const float* w2, const float* b2,
                       const float* lng, const float* lnb) {
    ln(src, lng, lnb, bln);
    gemm(bln, w1, b1, bhid, EMBD, 4 * EMBD);
    gelu_kernel<<<MTOK * 4 * EMBD / 256, 256, 0, stream>>>(bhid, MTOK * 4 * EMBD);
    gemm(bhid, w2, b2, bproj, 4 * EMBD, EMBD);
  };

  // experts 2,3: tf  (wqkv bqkv wo bo w1 b1 w2 b2 ln1g ln1b ln2g ln2b)
  for (int e = 2; e <= 3; ++e) {
    int p0 = 2 + (e - 2) * 12;
    attn_block(x, e, P(p0), P(p0 + 1), P(p0 + 2), P(p0 + 3), P(p0 + 8), P(p0 + 9));
    add_kernel<<<EW, 256, 0, stream>>>(bh, x, bproj);
    mlp_block(bh, P(p0 + 4), P(p0 + 5), P(p0 + 6), P(p0 + 7), P(p0 + 10), P(p0 + 11));
    accum_kernel<<<EW, 256, 0, stream>>>(y, we + e * MTOK, bh, bproj);
  }
  // experts 4,5: attn  (wqkv bqkv wo bo lng lnb)
  for (int e = 4; e <= 5; ++e) {
    int p0 = 26 + (e - 4) * 6;
    attn_block(x, e, P(p0), P(p0 + 1), P(p0 + 2), P(p0 + 3), P(p0 + 4), P(p0 + 5));
    accum_kernel<<<EW, 256, 0, stream>>>(y, we + e * MTOK, x, bproj);
  }
  // experts 6,7: mlp  (w1 b1 w2 b2 lng lnb)
  for (int e = 6; e <= 7; ++e) {
    int p0 = 38 + (e - 6) * 6;
    mlp_block(x, P(p0), P(p0 + 1), P(p0 + 2), P(p0 + 3), P(p0 + 4), P(p0 + 5));
    accum_kernel<<<EW, 256, 0, stream>>>(y, we + e * MTOK, x, bproj);
  }
}
